// GNT_68247030333626
// MI455X (gfx1250) — matrix operand order repaired and checked by reading, not compiled, not run on hardware
//
#include <hip/hip_runtime.h>

// Problem dims (fixed by the reference)
#define R_   256
#define S_   64
#define V_   8
#define D_   64
#define NT   (R_ * S_)        // 16384 ray-sample tokens
#define NTB  (R_ * S_ * V_)   // 131072 view tokens
#define HID_ 256

typedef __attribute__((ext_vector_type(16))) _Float16 v16h;
typedef __attribute__((ext_vector_type(8)))  float    v8f;

// ---------------------------------------------------------------------------
// Generic WMMA GEMM: C[ntok x M] = act(A[ntok x K] @ W[K x M] + bias) (+res)
// Block = 128 threads = 4 wave32s; 16 tokens x 64 cols per block.
// Both A-tile (16 x Kp) and W-tile (64 cols x Kp, transposed) staged in LDS
// as f16 with zero padding, so every fragment load is contiguous f16 runs
// (-> ds_load_b128 pairs) with no divergence in the K-loop.
// Fragment layouts per CDNA5 ISA 7.12.2 (16-bit A 16x32, B 32x16, f32 C/D).
// ---------------------------------------------------------------------------
__global__ __launch_bounds__(128) void gnt_gemm_wmma(
    const float* __restrict__ A, int lda,
    const float* __restrict__ W,          // [K, M] row-major
    const float* __restrict__ bias,       // [M] or null
    const float* __restrict__ res,        // residual [ntok, ldc] or null
    float* __restrict__ C, int ldc,
    int ntok, int K, int M, int relu)
{
  __shared__ _Float16 As[16 * 264];   // 16 rows, stride Kp+8 (<= 264)
  __shared__ _Float16 Wt[64 * 264];   // 64 cols (transposed), stride Kp+8
  const int Kp   = (K + 31) & ~31;
  const int KpA  = Kp + 8;            // padded row stride (keeps 16B align)
  const int tok0 = blockIdx.x << 4;
  const int cb   = blockIdx.y << 6;   // block's first output column
  const int tid  = threadIdx.x;

  // A-tile: 16 x Kp, fp32 -> f16, zero padded. Coalesced along k.
  for (int row = 0; row < 16; ++row) {
    int tok = tok0 + row;
    for (int k = tid; k < Kp; k += 128) {
      float v = 0.f;
      if (tok < ntok && k < K) v = A[(size_t)tok * lda + k];
      As[row * KpA + k] = (_Float16)v;
    }
  }
  // W-tile transposed: Wt[c][k] = W[k][cb+c]. Coalesced along columns.
  for (int idx = tid; idx < (Kp << 6); idx += 128) {
    int c = idx & 63;
    int k = idx >> 6;
    int col = cb + c;
    float v = 0.f;
    if (col < M && k < K) v = W[(size_t)k * M + col];
    Wt[c * KpA + k] = (_Float16)v;
  }
  __syncthreads();

  const int wave = tid >> 5;
  const int lane = tid & 31;
  const int h    = lane >> 4;       // half-wave select
  const int mn   = lane & 15;       // A: row M; B/D: col N
  const int col  = cb + (wave << 4) + mn;

  const _Float16* __restrict__ aBase = &As[mn * KpA + 8 * h];
  const _Float16* __restrict__ bBase = &Wt[((wave << 4) + mn) * KpA + 8 * h];

  v8f acc = {0.f, 0.f, 0.f, 0.f, 0.f, 0.f, 0.f, 0.f};
  for (int k0 = 0; k0 < Kp; k0 += 32) {
    v16h a, b;
    // A 16x32 f16: lanes0-15 K = k0+{0..7, 16..23}; lanes16-31 +8
#pragma unroll
    for (int e = 0; e < 8; ++e) {
      a[e]     = aBase[k0 + e];
      a[e + 8] = aBase[k0 + 16 + e];
    }
    // B 32x16 f16: lane -> N; K = k0+{0..7, 16..23}; lanes16-31 +8
#pragma unroll
    for (int e = 0; e < 8; ++e) {
      b[e]     = bBase[k0 + e];
      b[e + 8] = bBase[k0 + 16 + e];
    }
    acc = __builtin_amdgcn_wmma_f32_16x16x32_f16(
        false, a, false, b, (short)0, acc, false, false);
  }

  // D f32 16x16: VGPR r at lane -> C[M = r + 8*h][N = lane&15]
  const bool cok = (col < M);
#pragma unroll
  for (int r = 0; r < 8; ++r) {
    int tok = tok0 + r + 8 * h;
    if (tok < ntok && cok) {
      float v = acc[r];
      if (bias) v += bias[col];
      if (relu) v = v > 0.f ? v : 0.f;
      if (res)  v += res[(size_t)tok * ldc + col];
      C[(size_t)tok * ldc + col] = v;
    }
  }
}

// ---------------------------------------------------------------------------
// NeRF positional encoding of pts (per token) and normalized ray_d (per ray),
// written straight into the qfc concat buffer columns [64, 190).
// Layout per _embed: [x(3), {f=0..9: sin(x*2^f)(3), cos(x*2^f)(3)}] = 63
// ---------------------------------------------------------------------------
__global__ void embed_kernel(const float* __restrict__ pts,
                             const float* __restrict__ ray_d,
                             float* __restrict__ cat)
{
  int rs = blockIdx.x * blockDim.x + threadIdx.x;
  if (rs >= NT) return;
  int r = rs >> 6;
  float p0 = pts[rs * 3 + 0], p1 = pts[rs * 3 + 1], p2 = pts[rs * 3 + 2];
  float v0 = ray_d[r * 3 + 0], v1 = ray_d[r * 3 + 1], v2 = ray_d[r * 3 + 2];
  float inv = rsqrtf(v0 * v0 + v1 * v1 + v2 * v2);
  v0 *= inv; v1 *= inv; v2 *= inv;
  float* o = cat + (size_t)rs * 192 + 64;   // pts embed at o[0..62], views at o[63..125]
  o[0] = p0; o[1] = p1; o[2] = p2;
  o[63 + 0] = v0; o[63 + 1] = v1; o[63 + 2] = v2;
  float fr = 1.f;
  for (int f = 0; f < 10; ++f) {
    float a0 = p0 * fr, a1 = p1 * fr, a2 = p2 * fr;
    o[3 + f * 6 + 0] = sinf(a0); o[3 + f * 6 + 1] = sinf(a1); o[3 + f * 6 + 2] = sinf(a2);
    o[3 + f * 6 + 3] = cosf(a0); o[3 + f * 6 + 4] = cosf(a1); o[3 + f * 6 + 5] = cosf(a2);
    float b0 = v0 * fr, b1 = v1 * fr, b2 = v2 * fr;
    o[63 + 3 + f * 6 + 0] = sinf(b0); o[63 + 3 + f * 6 + 1] = sinf(b1); o[63 + 3 + f * 6 + 2] = sinf(b2);
    o[63 + 3 + f * 6 + 3] = cosf(b0); o[63 + 3 + f * 6 + 4] = cosf(b1); o[63 + 3 + f * 6 + 5] = cosf(b2);
    fr *= 2.f;
  }
}

// q[rs,d] = max over views of x[rs,v,d]
__global__ void maxpool_kernel(const float* __restrict__ x, float* __restrict__ q)
{
  int idx = blockIdx.x * blockDim.x + threadIdx.x;
  if (idx >= NT * 64) return;
  int rs = idx >> 6, d = idx & 63;
  size_t base = (size_t)rs * (V_ * 64) + d;
  float m = x[base];
  for (int v = 1; v < V_; ++v) m = fmaxf(m, x[base + v * 64]);
  q[idx] = m;
}

// LayerNorm over last dim (64). One 64-thread block per token.
__global__ __launch_bounds__(64) void ln_kernel(
    const float* __restrict__ x, const float* __restrict__ g,
    const float* __restrict__ b, float* __restrict__ y)
{
  int tok = blockIdx.x;
  int d   = threadIdx.x;
  float v = x[(size_t)tok * 64 + d];
  __shared__ float red[64];
  red[d] = v; __syncthreads();
  for (int o = 32; o > 0; o >>= 1) { if (d < o) red[d] += red[d + o]; __syncthreads(); }
  float mean = red[0] * (1.f / 64.f);
  __syncthreads();
  float c = v - mean;
  red[d] = c * c; __syncthreads();
  for (int o = 32; o > 0; o >>= 1) { if (d < o) red[d] += red[d + o]; __syncthreads(); }
  float var = red[0] * (1.f / 64.f);
  y[(size_t)tok * 64 + d] = c * rsqrtf(var + 1e-6f) * g[d] + b[d];
}

// kq = kk - qq(broadcast over views) + pe ;  vpe = vv + pe
__global__ void fuse_kernel(float* __restrict__ kq, float* __restrict__ vpe,
                            const float* __restrict__ pe, const float* __restrict__ qq)
{
  int idx = blockIdx.x * blockDim.x + threadIdx.x;
  if (idx >= NTB * 64) return;
  int d  = idx & 63;
  int rs = idx >> 9;            // / (V_*64)
  float p = pe[idx];
  kq[idx]  = kq[idx] - qq[(size_t)rs * 64 + d] + p;
  vpe[idx] = vpe[idx] + p;
}

// Masked softmax over the 8 views, then weighted sum of (v+pe).
__global__ void attn_reduce_kernel(const float* __restrict__ a,
                                   const float* __restrict__ vpe,
                                   const int* __restrict__ mask,
                                   float* __restrict__ out)
{
  int idx = blockIdx.x * blockDim.x + threadIdx.x;
  if (idx >= NT * 64) return;
  int rs = idx >> 6, d = idx & 63;
  size_t base = (size_t)rs * (V_ * 64) + d;
  float av[V_];
  float mx = -3.0e38f;
  for (int v = 0; v < V_; ++v) {
    float t = a[base + v * 64];
    if (mask[rs * V_ + v] == 0) t = -1e9f;
    av[v] = t;
    mx = fmaxf(mx, t);
  }
  float sum = 0.f;
  for (int v = 0; v < V_; ++v) { av[v] = __expf(av[v] - mx); sum += av[v]; }
  float acc = 0.f;
  for (int v = 0; v < V_; ++v) acc += av[v] * vpe[base + v * 64];
  out[idx] = acc / sum;
}

// Copy current q into concat buffer columns [0,64)
__global__ void copyq_kernel(const float* __restrict__ q, float* __restrict__ cat)
{
  int idx = blockIdx.x * blockDim.x + threadIdx.x;
  if (idx >= NT * 64) return;
  int rs = idx >> 6, d = idx & 63;
  cat[(size_t)rs * 192 + d] = q[idx];
}

// Multi-head self-attention along the 64 samples of each ray.
// Thread per (ray, head, query): online softmax over 64 keys, hd = 16.
__global__ void self_attn_kernel(const float* __restrict__ q,
                                 const float* __restrict__ k,
                                 const float* __restrict__ v,
                                 float* __restrict__ o)
{
  int idx = blockIdx.x * blockDim.x + threadIdx.x;
  if (idx >= R_ * 4 * 64) return;
  int r  = idx >> 8;
  int h  = (idx >> 6) & 3;
  int qi = idx & 63;
  const float* qp = q + ((size_t)(r * 64 + qi)) * 64 + h * 16;
  float qv[16];
#pragma unroll
  for (int t = 0; t < 16; ++t) qv[t] = qp[t];
  float m = -3.0e38f, l = 0.f, acc[16];
#pragma unroll
  for (int t = 0; t < 16; ++t) acc[t] = 0.f;
  for (int kj = 0; kj < 64; ++kj) {
    const float* kp = k + ((size_t)(r * 64 + kj)) * 64 + h * 16;
    float s = 0.f;
#pragma unroll
    for (int t = 0; t < 16; ++t) s += qv[t] * kp[t];
    s *= 0.25f;                         // 1/sqrt(16)
    float mn  = fmaxf(m, s);
    float cor = __expf(m - mn);
    float w   = __expf(s - mn);
    l = l * cor + w;
    const float* vp = v + ((size_t)(r * 64 + kj)) * 64 + h * 16;
#pragma unroll
    for (int t = 0; t < 16; ++t) acc[t] = acc[t] * cor + w * vp[t];
    m = mn;
  }
  float* op  = o + ((size_t)(r * 64 + qi)) * 64 + h * 16;
  float invl = 1.f / l;
#pragma unroll
  for (int t = 0; t < 16; ++t) op[t] = acc[t] * invl;
}

// Mean over samples then 64->3 projection. One 64-thread block per ray.
__global__ __launch_bounds__(64) void mean_rgb_kernel(
    const float* __restrict__ ln, const float* __restrict__ w,
    const float* __restrict__ b, float* __restrict__ out)
{
  int r = blockIdx.x;
  int d = threadIdx.x;
  float s = 0.f;
  for (int ss = 0; ss < S_; ++ss) s += ln[((size_t)(r * S_ + ss)) * 64 + d];
  __shared__ float hm[64];
  hm[d] = s * (1.f / (float)S_);
  __syncthreads();
  if (d < 3) {
    float acc = b[d];
    for (int j = 0; j < 64; ++j) acc += hm[j] * w[j * 3 + d];
    out[r * 3 + d] = acc;
  }
}

// ---------------------------------------------------------------------------
static inline void launch_gemm(hipStream_t st, const float* A, int lda,
                               const float* W, const float* bias,
                               const float* res, float* C, int ldc,
                               int ntok, int K, int M, int relu)
{
  dim3 g((ntok + 15) / 16, (M + 63) / 64);
  gnt_gemm_wmma<<<g, 128, 0, st>>>(A, lda, W, bias, res, C, ldc, ntok, K, M, relu);
}

extern "C" void kernel_launch(void* const* d_in, const int* in_sizes, int n_in,
                              void* d_out, int out_size, void* d_ws, size_t ws_size,
                              hipStream_t stream)
{
  (void)in_sizes; (void)n_in; (void)out_size; (void)ws_size;

  // Inputs in setup_inputs() insertion order; params pytree flattened
  // recursively in insertion order (qfc=None contributes no leaves).
  const float* rgb_feat = (const float*)d_in[0];
  const float* ray_diff = (const float*)d_in[1];
  const int*   mask     = (const int*)d_in[2];
  const float* pts      = (const float*)d_in[3];
  const float* ray_d    = (const float*)d_in[4];
  int pi = 5;
  auto nf = [&]() { return (const float*)d_in[pi++]; };

  const float* rf0w = nf(); const float* rf0b = nf();
  const float* rf1w = nf(); const float* rf1b = nf();
  const float* norm_g = nf(); const float* norm_b = nf();
  const float* rgbw = nf(); const float* rgbb = nf();

  struct CrossP { const float *ang,*anb,*fng,*fnb,*qw,*kw,*vw,
                  *p0w,*p0b,*p1w,*p1b,*a0w,*a0b,*a1w,*a1b,*ow,*ob,
                  *f0w,*f0b,*f1w,*f1b; };
  struct SelfP  { const float *ang,*anb,*fng,*fnb,*qw,*kw,*vw,*ow,*ob,
                  *f0w,*f0b,*f1w,*f1b; };
  struct QfcP   { const float *w0,*b0,*w1,*b1; };
  CrossP cr[4]; SelfP sf[4]; QfcP qf[4] = {};

  for (int i = 0; i < 4; ++i) {
    CrossP& c = cr[i];
    c.ang = nf(); c.anb = nf(); c.fng = nf(); c.fnb = nf();
    c.qw = nf(); c.kw = nf(); c.vw = nf();
    c.p0w = nf(); c.p0b = nf(); c.p1w = nf(); c.p1b = nf();
    c.a0w = nf(); c.a0b = nf(); c.a1w = nf(); c.a1b = nf();
    c.ow = nf(); c.ob = nf();
    c.f0w = nf(); c.f0b = nf(); c.f1w = nf(); c.f1b = nf();
    SelfP& s = sf[i];
    s.ang = nf(); s.anb = nf(); s.fng = nf(); s.fnb = nf();
    s.qw = nf(); s.kw = nf(); s.vw = nf();
    s.ow = nf(); s.ob = nf();
    s.f0w = nf(); s.f0b = nf(); s.f1w = nf(); s.f1b = nf();
    if (i % 2 == 0) { qf[i].w0 = nf(); qf[i].b0 = nf(); qf[i].w1 = nf(); qf[i].b1 = nf(); }
  }

  // Workspace carve-up (~184 MB; everything L2-resident on 192MB L2)
  char* ws = (char*)d_ws;
  size_t off = 0;
  auto alloc = [&](size_t bytes) {
    float* p = (float*)(ws + off);
    off += (bytes + 255) & ~(size_t)255;
    return p;
  };
  float* X   = alloc((size_t)NTB * 64 * 4);   // rgbfeat features (kv source)
  float* T0  = alloc((size_t)NTB * 64 * 4);   // kk / attn input; reused by self-attn
  float* T1  = alloc((size_t)NTB * 64 * 4);   // vv -> v+pe
  float* T2  = alloc((size_t)NTB * 64 * 4);   // pe -> attn logits
  float* H8  = alloc((size_t)NTB * 8 * 4);    // 8-wide MLP hidden
  float* Q   = alloc((size_t)NT * 64 * 4);    // running q
  float* QQ  = alloc((size_t)NT * 64 * 4);    // q projection
  float* LNB = alloc((size_t)NT * 64 * 4);    // layernorm / temp
  float* ATT = alloc((size_t)NT * 64 * 4);    // cross-attn output
  float* FFH = alloc((size_t)NT * 256 * 4);   // FF hidden
  float* CAT = alloc((size_t)NT * 192 * 4);   // [q(64) | pts_pe(63) | view_pe(63)]
  float* SQb = T0;
  float* SKb = T0 + (size_t)NT * 64;
  float* SVb = T0 + (size_t)2 * NT * 64;
  float* SOb = T0 + (size_t)3 * NT * 64;

  // Positional encodings (constant across layers) -> CAT[:,64:190]
  embed_kernel<<<(NT + 255) / 256, 256, 0, stream>>>(pts, ray_d, CAT);

  // rgbfeat MLP: [NTB,35] -> ReLU 64 -> 64
  launch_gemm(stream, rgb_feat, 35, rf0w, rf0b, nullptr, T0, 64, NTB, 35, 64, 1);
  launch_gemm(stream, T0, 64, rf1w, rf1b, nullptr, X, 64, NTB, 64, 64, 0);
  // q = max over views
  maxpool_kernel<<<(NT * 64 + 255) / 256, 256, 0, stream>>>(X, Q);

  for (int i = 0; i < 4; ++i) {
    const CrossP& c = cr[i];
    // ---- cross attention over views ----
    ln_kernel<<<NT, 64, 0, stream>>>(Q, c.ang, c.anb, LNB);
    launch_gemm(stream, LNB, 64, c.qw, nullptr, nullptr, QQ, 64, NT, 64, 64, 0);
    launch_gemm(stream, X, 64, c.kw, nullptr, nullptr, T0, 64, NTB, 64, 64, 0);
    launch_gemm(stream, X, 64, c.vw, nullptr, nullptr, T1, 64, NTB, 64, 64, 0);
    launch_gemm(stream, ray_diff, 4, c.p0w, c.p0b, nullptr, H8, 8, NTB, 4, 8, 1);
    launch_gemm(stream, H8, 8, c.p1w, c.p1b, nullptr, T2, 64, NTB, 8, 64, 0);
    fuse_kernel<<<(NTB * 64 + 255) / 256, 256, 0, stream>>>(T0, T1, T2, QQ);
    launch_gemm(stream, T0, 64, c.a0w, c.a0b, nullptr, H8, 8, NTB, 64, 8, 1);
    launch_gemm(stream, H8, 8, c.a1w, c.a1b, nullptr, T2, 64, NTB, 8, 64, 0);
    attn_reduce_kernel<<<(NT * 64 + 255) / 256, 256, 0, stream>>>(T2, T1, mask, ATT);
    launch_gemm(stream, ATT, 64, c.ow, c.ob, Q, Q, 64, NT, 64, 64, 0);
    ln_kernel<<<NT, 64, 0, stream>>>(Q, c.fng, c.fnb, LNB);
    launch_gemm(stream, LNB, 64, c.f0w, c.f0b, nullptr, FFH, 256, NT, 64, 256, 1);
    launch_gemm(stream, FFH, 256, c.f1w, c.f1b, Q, Q, 64, NT, 256, 64, 0);

    // ---- qfc on even layers: q = MLP(concat(q, pts_pe, view_pe)) ----
    if (i % 2 == 0) {
      copyq_kernel<<<(NT * 64 + 255) / 256, 256, 0, stream>>>(Q, CAT);
      launch_gemm(stream, CAT, 192, qf[i].w0, qf[i].b0, nullptr, LNB, 64, NT, 190, 64, 1);
      launch_gemm(stream, LNB, 64, qf[i].w1, qf[i].b1, nullptr, Q, 64, NT, 64, 64, 0);
    }

    // ---- self attention along samples ----
    const SelfP& s = sf[i];
    ln_kernel<<<NT, 64, 0, stream>>>(Q, s.ang, s.anb, LNB);
    launch_gemm(stream, LNB, 64, s.qw, nullptr, nullptr, SQb, 64, NT, 64, 64, 0);
    launch_gemm(stream, LNB, 64, s.kw, nullptr, nullptr, SKb, 64, NT, 64, 64, 0);
    launch_gemm(stream, LNB, 64, s.vw, nullptr, nullptr, SVb, 64, NT, 64, 64, 0);
    self_attn_kernel<<<(R_ * 4 * 64 + 255) / 256, 256, 0, stream>>>(SQb, SKb, SVb, SOb);
    launch_gemm(stream, SOb, 64, s.ow, s.ob, Q, Q, 64, NT, 64, 64, 0);
    ln_kernel<<<NT, 64, 0, stream>>>(Q, s.fng, s.fnb, LNB);
    launch_gemm(stream, LNB, 64, s.f0w, s.f0b, nullptr, FFH, 256, NT, 64, 256, 1);
    launch_gemm(stream, FFH, 256, s.f1w, s.f1b, Q, Q, 64, NT, 256, 64, 0);
  }

  // Final norm, mean over samples, rgb projection
  ln_kernel<<<NT, 64, 0, stream>>>(Q, norm_g, norm_b, LNB);
  mean_rgb_kernel<<<R_, 64, 0, stream>>>(LNB, rgbw, rgbb, (float*)d_out);
}
